// SRU_19164144075359
// MI455X (gfx1250) — matrix operand order repaired and checked by reading, not compiled, not run on hardware
//
#include <hip/hip_runtime.h>
#include <hip/hip_bf16.h>

// ---------------------------------------------------------------------------
// SRU text classifier for MI455X (gfx1250, wave32, WMMA).
// S=2048 B=32 V=50000 D=H=512 L=2 C=10, N3 = 3*H = 1536, M = S*B = 65536.
//
// Heavy op: per-layer GEMM (65536 x 512) @ (512 x 1536) via
// v_wmma_f32_16x16x32_bf16 (f32 accumulate), software-pipelined one K-step
// deep with sched_barriers so fragment loads overlap WMMAs. Both operands
// live in bf16 (weights 3MB, activation shadow 67MB -> L2-resident @192MB).
// Scan + highway are fused into one pass over U.
// ---------------------------------------------------------------------------

#define SEQ   2048
#define BATCH 32
#define DIM   512
#define HID   512
#define N3    1536
#define NCLS  10
#define MROWS (SEQ * BATCH)   // 65536

typedef __attribute__((ext_vector_type(16))) __bf16       bf16x16;
typedef __attribute__((ext_vector_type(8)))  float        f32x8;
typedef __attribute__((ext_vector_type(4)))  unsigned int u32x4;

union FragAB {
    bf16x16        v;
    u32x4          q[2];
    unsigned short s[16];
};

#if __has_builtin(__builtin_amdgcn_sched_barrier)
#define SCHED_FENCE() __builtin_amdgcn_sched_barrier(0)
#else
#define SCHED_FENCE()
#endif

__device__ __forceinline__ unsigned short f2bf(float f) {
    // round-to-nearest-even f32 -> bf16 (NaN edge ignored; inputs are tame)
    unsigned int u = __float_as_uint(f);
    return (unsigned short)((u + 0x7FFFu + ((u >> 16) & 1u)) >> 16);
}

__device__ __forceinline__ unsigned int pack2bf(float lo, float hi) {
    return (unsigned int)f2bf(lo) | ((unsigned int)f2bf(hi) << 16);
}

__device__ __forceinline__ float sigmoidf_dev(float x) {
    return 1.0f / (1.0f + __expf(-x));
}

// ---------------------------------------------------------------------------
// 1) Weights: (L, 512, 1536) f32 row-major -> (L, 1536, 512) bf16 N-major
// ---------------------------------------------------------------------------
__global__ void wt_convert_kernel(const float* __restrict__ W,
                                  unsigned short* __restrict__ Wt) {
    size_t idx = (size_t)blockIdx.x * blockDim.x + threadIdx.x;
    if (idx >= (size_t)2 * N3 * DIM) return;
    size_t l   = idx / ((size_t)N3 * DIM);
    size_t rem = idx % ((size_t)N3 * DIM);
    size_t n   = rem / DIM;
    size_t k   = rem % DIM;
    float v = W[l * (size_t)DIM * N3 + k * N3 + n];
    Wt[idx] = f2bf(v);
}

// ---------------------------------------------------------------------------
// 2) Embedding gather: h[row, :] = embed[x[row], :]; also bf16 shadow.
// ---------------------------------------------------------------------------
__global__ void embed_gather_kernel(const int* __restrict__ x,
                                    const float* __restrict__ embed,
                                    float* __restrict__ h,
                                    unsigned short* __restrict__ hbf) {
    const int row = blockIdx.x;
    const int tok = x[row];
    const float4* src = (const float4*)(embed + (size_t)tok * DIM);
    float4 v = src[threadIdx.x];                 // 128 thr * 4 floats = 512
    ((float4*)(h + (size_t)row * DIM))[threadIdx.x] = v;
    uint2 p;
    p.x = pack2bf(v.x, v.y);
    p.y = pack2bf(v.z, v.w);
    ((uint2*)(hbf + (size_t)row * DIM))[threadIdx.x] = p;
}

// ---------------------------------------------------------------------------
// 3) GEMM + gate epilogue, software-pipelined.
//    U[m, 0:512]      = raw x_tilde
//    U[m, 512:1024]   = sigmoid(u + b_f)
//    U[m, 1024:1536]  = sigmoid(u + b_r)
//    Block: 256 thr = 8 waves. Wave w: rows [bx*128 + w*16, +16),
//    cols [by*96, +96) -> 6 accum tiles, K stepped by 32 (bf16).
//    All loads: uniform SGPR base + 32-bit lane offset (saddr/voffset form).
// ---------------------------------------------------------------------------
__device__ __forceinline__ void load_a_frag(FragAB& a,
                                            const unsigned short* __restrict__ base,
                                            unsigned off) {
    // A 16x32 bf16 ISA layout: lanes 0-15 row m K=+{0..7},{16..23};
    // lanes 16-31 row m K=+8+{0..7}, +24+{0..7} (laneHi folded into off)
    a.q[0] = *(const u32x4*)(base + off);        // 8 bf16 at K offset  +0
    a.q[1] = *(const u32x4*)(base + off + 16);   // 8 bf16 at K offset +16
}

__device__ __forceinline__ void load_b_frag(FragAB& b,
                                            const unsigned short* __restrict__ base,
                                            unsigned off) {
    // B 32x16 bf16 layout: lanes 0-15 col n K=+{0..7},{16..23};
    //                      lanes 16-31 col n K=+8+{0..7}, +24+{0..7} (laneHi folded into off)
    b.q[0] = *(const u32x4*)(base + off);
    b.q[1] = *(const u32x4*)(base + off + 16);
}

__global__ __launch_bounds__(256, 1)
void sru_gemm_kernel(const unsigned short* __restrict__ Abf,  // (M, 512) bf16
                     const unsigned short* __restrict__ Wt,   // (1536, 512) bf16 N-major
                     const float* __restrict__ bias,          // (1024,) f32
                     float* __restrict__ U)                   // (M, 1536) f32
{
    const int wave   = threadIdx.x >> 5;
    const int lane   = threadIdx.x & 31;
    const int laneLo = lane & 15;
    const int laneHi = lane >> 4;       // 0 or 1

    const int rowBase = blockIdx.x * 128 + wave * 16;
    const int colBase = blockIdx.y * 96;
    const int m       = rowBase + laneLo;

    // 32-bit lane offsets (A <= 33.5M elems, W <= 1.57M elems: both fit u32)
    const unsigned aOff = (unsigned)m * DIM + laneHi * 8;
    unsigned bOff[6];
    #pragma unroll
    for (int t = 0; t < 6; ++t)
        bOff[t] = (unsigned)(colBase + t * 16 + laneLo) * DIM + laneHi * 8;

    f32x8 acc[6];
    #pragma unroll
    for (int t = 0; t < 6; ++t) acc[t] = (f32x8){};

    // ---- software pipeline: fragments for k+32 load under k's WMMAs ----
    FragAB aCur, aNxt;
    FragAB bCur[6], bNxt[6];

    load_a_frag(aCur, Abf, aOff);
    #pragma unroll
    for (int t = 0; t < 6; ++t) load_b_frag(bCur[t], Wt, bOff[t]);

    #pragma unroll
    for (int k0 = 0; k0 < DIM; k0 += 32) {
        const int k1 = k0 + 32;
        if (k1 < DIM) {
            load_a_frag(aNxt, Abf, aOff + k1);
            #pragma unroll
            for (int t = 0; t < 6; ++t) load_b_frag(bNxt[t], Wt, bOff[t] + k1);
        }
        SCHED_FENCE();   // loads above may not sink below the WMMAs
        #pragma unroll
        for (int t = 0; t < 6; ++t)
            acc[t] = __builtin_amdgcn_wmma_f32_16x16x32_bf16(
                false, aCur.v, false, bCur[t].v, (short)0, acc[t], false, false);
        SCHED_FENCE();   // WMMAs may not sink into next iteration's loads
        if (k1 < DIM) {
            aCur = aNxt;
            #pragma unroll
            for (int t = 0; t < 6; ++t) bCur[t] = bNxt[t];
        }
    }

    // ---- epilogue: C/D layout VGPR r -> row rowBase + r + 8*laneHi, col laneLo
    #pragma unroll
    for (int t = 0; t < 6; ++t) {
        const int n = colBase + t * 16 + laneLo;
        const bool gate = (n >= HID);
        const float bb  = gate ? bias[n - HID] : 0.0f;
        #pragma unroll
        for (int r = 0; r < 8; ++r) {
            const int row = rowBase + r + laneHi * 8;
            float v = acc[t][r];
            if (gate) v = sigmoidf_dev(v + bb);
            U[(size_t)row * N3 + n] = v;
        }
    }
}

// ---------------------------------------------------------------------------
// 4) Fused light recurrence + highway. One thread per (b, h):
//      c_t = f_t*c + (1-f_t)*x_t
//      h_t = r_t*tanh(c_t) + (1-r_t)*h_in_t      (in-place on h)
//    Also refreshes the bf16 shadow (next layer's GEMM A operand).
// ---------------------------------------------------------------------------
__global__ __launch_bounds__(256)
void scan_highway_kernel(const float* __restrict__ U,
                         float* __restrict__ h,
                         unsigned short* __restrict__ hbf) {
    const int idx = blockIdx.x * blockDim.x + threadIdx.x; // 0..B*H-1
    const int b  = idx >> 9;     // / 512
    const int hh = idx & 511;
    float c = 0.0f;
    for (int s = 0; s < SEQ; ++s) {
        const size_t row = (size_t)s * BATCH + b;
        const float* p = U + row * N3;
        const float xt = p[hh];
        const float f  = p[HID + hh];
        const float r  = p[2 * HID + hh];
        c = f * c + (1.0f - f) * xt;
        const size_t o = row * DIM + hh;
        const float hin = h[o];
        const float ho  = r * tanhf(c) + (1.0f - r) * hin;
        h[o]   = ho;
        hbf[o] = f2bf(ho);
    }
}

// ---------------------------------------------------------------------------
// 5) tanh -> maxpool over S -> tanh -> @Wc + bc. tanh is monotone, so
//    max_s tanh(h) == tanh(max_s h). One block per batch element.
// ---------------------------------------------------------------------------
__global__ __launch_bounds__(256)
void final_kernel(const float* __restrict__ h,
                  const float* __restrict__ Wc,
                  const float* __restrict__ bc,
                  float* __restrict__ out) {
    __shared__ float p[HID];
    const int b = blockIdx.x;
    for (int hh = threadIdx.x; hh < HID; hh += blockDim.x) {
        float m = -3.4e38f;
        for (int s = 0; s < SEQ; ++s)
            m = fmaxf(m, h[((size_t)s * BATCH + b) * HID + hh]);
        p[hh] = tanhf(tanhf(m));
    }
    __syncthreads();
    if (threadIdx.x < NCLS) {
        float acc = bc[threadIdx.x];
        for (int hh = 0; hh < HID; ++hh)
            acc += p[hh] * Wc[hh * NCLS + threadIdx.x];
        out[b * NCLS + threadIdx.x] = acc;
    }
}

// ---------------------------------------------------------------------------
// Launch
// ---------------------------------------------------------------------------
extern "C" void kernel_launch(void* const* d_in, const int* in_sizes, int n_in,
                              void* d_out, int out_size, void* d_ws, size_t ws_size,
                              hipStream_t stream) {
    const int*   x     = (const int*)d_in[0];
    const float* embed = (const float*)d_in[1];
    const float* W     = (const float*)d_in[2];
    const float* b     = (const float*)d_in[3];
    const float* Wc    = (const float*)d_in[4];
    const float* bc    = (const float*)d_in[5];
    float*       out   = (float*)d_out;

    // workspace carve-up (~580 MB):
    //   h   : M*512  f32       = 134,217,728 B   @ 0
    //   U   : M*1536 f32       = 402,653,184 B   @ 134,217,728
    //   Wt  : L*1536*512 bf16  =   3,145,728 B   @ 536,870,912
    //   hbf : M*512  bf16      =  67,108,864 B   @ 540,016,640
    char* ws = (char*)d_ws;
    float*          hbuf = (float*)ws;
    float*          U    = (float*)(ws + 134217728ull);
    unsigned short* Wt   = (unsigned short*)(ws + 536870912ull);
    unsigned short* hbf  = (unsigned short*)(ws + 540016640ull);

    // weights -> bf16 N-major (cheap; deterministic every call)
    {
        const size_t tot = (size_t)2 * N3 * DIM;
        wt_convert_kernel<<<(unsigned)((tot + 255) / 256), 256, 0, stream>>>(W, Wt);
    }

    // embedding gather (f32 + bf16 shadow)
    embed_gather_kernel<<<MROWS, 128, 0, stream>>>(x, embed, hbuf, hbf);

    // two SRU layers
    for (int l = 0; l < 2; ++l) {
        dim3 grid(MROWS / 128, N3 / 96);   // (512, 16)
        sru_gemm_kernel<<<grid, 256, 0, stream>>>(
            hbf, Wt + (size_t)l * N3 * DIM, b + (size_t)l * 2 * HID, U);
        scan_highway_kernel<<<(BATCH * HID) / 256, 256, 0, stream>>>(U, hbuf, hbf);
    }

    // pooled classifier
    final_kernel<<<BATCH, 256, 0, stream>>>(hbuf, Wc, bc, out);
}
